// MixtralSparseMoe_48455821033858
// MI455X (gfx1250) — compile-verified, matrix operand order checked by reading, not run on hardware
//
#include <hip/hip_runtime.h>

// ---------------------------------------------------------------------------
// Mixtral-style sparse MoE (top-2 of 8 experts), MI455X / gfx1250.
//   K1 router  -> logits (to d_out tail) + per-expert token buckets
//   K2 offsets -> prefix sum of bucket counts
//   K3 gemm1   -> h = silu(x@w1)*(x@w3)  (f32 WMMA 16x16x4), h stored bf16,
//                 async global->LDS double buffering
//   K4 gemm2   -> out += weight * (h @ w2)  (bf16 WMMA 16x16x32, f32 acc)
// ---------------------------------------------------------------------------

typedef __attribute__((ext_vector_type(2)))  float  v2f;
typedef __attribute__((ext_vector_type(8)))  float  v8f;
typedef __attribute__((ext_vector_type(16))) __bf16 v16bf;
typedef int v4i_ __attribute__((vector_size(16)));   // matches async-builtin param

union BfFrag { v16bf v; unsigned int u[8]; };

constexpr int TT = 8192;   // tokens = B*S
constexpr int DD = 1024;   // hidden dim
constexpr int FF = 3584;   // ffn dim
constexpr int EE = 8;      // experts

#if __has_builtin(__builtin_amdgcn_global_load_async_to_lds_b128)
#define MOE_ASYNC 1
#else
#define MOE_ASYNC 0
#endif

__device__ __forceinline__ unsigned short f32_to_bf16(float f) {
  unsigned int u = __float_as_uint(f);
  u += 0x7FFFu + ((u >> 16) & 1u);   // round-to-nearest-even
  return (unsigned short)(u >> 16);
}

#if MOE_ASYNC
__device__ __forceinline__ void async_cp16(const float* g, void* l) {
  __builtin_amdgcn_global_load_async_to_lds_b128(
      (v4i_*)g, (v4i_*)l, 0, 0);
}
__device__ __forceinline__ void wait_async0() {
#if __has_builtin(__builtin_amdgcn_s_wait_asynccnt)
  __builtin_amdgcn_s_wait_asynccnt(0);
#else
  asm volatile("s_wait_asynccnt 0x0" ::: "memory");
#endif
}
#endif

// ---------------------------------------------------------------------------
// K1: router. One thread per token; gate_w cached in LDS (32 KB).
// ---------------------------------------------------------------------------
__global__ __launch_bounds__(256) void moe_router(
    const float* __restrict__ x, const float* __restrict__ gw,
    float* __restrict__ logits_out, int* __restrict__ cnt,
    int* __restrict__ btok, float* __restrict__ bwt)
{
  __shared__ float gsh[EE][DD];
  const int tid = threadIdx.x;
  for (int i = tid; i < EE * DD; i += 256)
    gsh[i >> 10][i & (DD - 1)] = gw[i];
  __syncthreads();

  const int t = blockIdx.x * 256 + tid;
  float acc[EE];
#pragma unroll
  for (int e = 0; e < EE; ++e) acc[e] = 0.f;

  const float4* xr = reinterpret_cast<const float4*>(x + (size_t)t * DD);
  for (int i = 0; i < DD / 4; ++i) {
    float4 xv = xr[i];
    int d = i * 4;
#pragma unroll
    for (int e = 0; e < EE; ++e) {
      acc[e] += xv.x * gsh[e][d]     + xv.y * gsh[e][d + 1]
              + xv.z * gsh[e][d + 2] + xv.w * gsh[e][d + 3];
    }
  }

#pragma unroll
  for (int e = 0; e < EE; ++e) logits_out[(size_t)t * EE + e] = acc[e];

  float mx = acc[0];
#pragma unroll
  for (int e = 1; e < EE; ++e) mx = fmaxf(mx, acc[e]);
  float p[EE];
#pragma unroll
  for (int e = 0; e < EE; ++e) p[e] = __expf(acc[e] - mx);

  int i1 = 0; float p1 = p[0];
#pragma unroll
  for (int e = 1; e < EE; ++e) if (p[e] > p1) { p1 = p[e]; i1 = e; }
  int i2 = -1; float p2 = -1.f;
#pragma unroll
  for (int e = 0; e < EE; ++e) if (e != i1 && p[e] > p2) { p2 = p[e]; i2 = e; }
  const float inv = 1.f / (p1 + p2);

  int pos1 = atomicAdd(&cnt[i1], 1);
  btok[i1 * TT + pos1] = t;  bwt[i1 * TT + pos1] = p1 * inv;
  int pos2 = atomicAdd(&cnt[i2], 1);
  btok[i2 * TT + pos2] = t;  bwt[i2 * TT + pos2] = p2 * inv;
}

// ---------------------------------------------------------------------------
// K2: prefix sum of bucket counts
// ---------------------------------------------------------------------------
__global__ void moe_offsets(const int* __restrict__ cnt, int* __restrict__ offs)
{
  if (threadIdx.x == 0) {
    int a = 0;
    for (int e = 0; e < EE; ++e) { offs[e] = a; a += cnt[e]; }
  }
}

// ---------------------------------------------------------------------------
// shared WMMA inner step for gemm1 (32-wide K chunk, f32 WMMA K=4)
// ---------------------------------------------------------------------------
__device__ __forceinline__ void gemm1_compute(
    const float (*__restrict__ As)[36], const float (*__restrict__ B1)[72],
    const float (*__restrict__ B3)[72],
    int wm, int wn, int mf, int kh, v8f* acc1, v8f* acc3)
{
#pragma unroll
  for (int ks = 0; ks < 32; ks += 4) {
    v2f a = { As[wm + mf][ks + kh], As[wm + mf][ks + kh + 1] };
#pragma unroll
    for (int s = 0; s < 2; ++s) {
      v2f b1 = { B1[ks + kh][wn + s * 16 + mf], B1[ks + kh + 1][wn + s * 16 + mf] };
      v2f b3 = { B3[ks + kh][wn + s * 16 + mf], B3[ks + kh + 1][wn + s * 16 + mf] };
      acc1[s] = __builtin_amdgcn_wmma_f32_16x16x4_f32(
          false, a, false, b1, (short)0, acc1[s], false, false);
      acc3[s] = __builtin_amdgcn_wmma_f32_16x16x4_f32(
          false, a, false, b3, (short)0, acc3[s], false, false);
    }
  }
}

// ---------------------------------------------------------------------------
// K3: grouped GEMM1 — h = silu(Xg @ w1[e]) * (Xg @ w3[e]), stored bf16.
// 64x64 tile, 8 waves (4Mx2N), K-chunk 32, async double-buffered LDS.
// ---------------------------------------------------------------------------
__global__ __launch_bounds__(256) void moe_gemm1(
    const float* __restrict__ x,  const float* __restrict__ w1,
    const float* __restrict__ w3, const int* __restrict__ cnt,
    const int* __restrict__ offs, const int* __restrict__ btok,
    unsigned short* __restrict__ hbuf)
{
  const int e     = blockIdx.z;
  const int count = cnt[e];
  const int m0    = blockIdx.y * 64;
  if (m0 >= count) return;
  const int f0    = blockIdx.x * 64;

#if MOE_ASYNC
  __shared__ float As [2][64][36];
  __shared__ float Bs1[2][32][72];
  __shared__ float Bs3[2][32][72];
#else
  __shared__ float As [1][64][36];
  __shared__ float Bs1[1][32][72];
  __shared__ float Bs3[1][32][72];
#endif
  __shared__ int toks[64];

  const int tid  = threadIdx.x;
  const int lane = tid & 31, wid = tid >> 5;
  const int wm = (wid >> 1) * 16;       // 0,16,32,48
  const int wn = (wid & 1) * 32;        // 0,32
  const int mf = lane & 15;
  const int kh = (lane >> 4) << 1;      // f32 WMMA: lanes 0-15 K0,1; 16-31 K2,3

  if (tid < 64) {
    int pos = m0 + tid;
    toks[tid] = btok[e * TT + (pos < count ? pos : count - 1)];
  }
  __syncthreads();

  // per-thread fill assignments (16B granules)
  const int rA = tid >> 3, cA = (tid & 7) * 4;    // A: 64 rows x 8 float4
  const int rB = tid >> 4, cB = (tid & 15) * 4;   // B: 32 rows x 16 float4
  const float* w1p = w1 + (size_t)e * DD * FF;
  const float* w3p = w3 + (size_t)e * DD * FF;
  const float* xr0 = x + (size_t)toks[rA] * DD + cA;
  const float* xr1 = x + (size_t)toks[rA + 32] * DD + cA;
  const float* w1b = w1p + (size_t)rB * FF + f0 + cB;
  const float* w3b = w3p + (size_t)rB * FF + f0 + cB;

  const v8f zero8 = {0.f,0.f,0.f,0.f,0.f,0.f,0.f,0.f};
  v8f acc1[2] = {zero8, zero8};
  v8f acc3[2] = {zero8, zero8};

#if MOE_ASYNC
  // issue chunk 0
  {
    async_cp16(xr0, &As[0][rA][cA]);
    async_cp16(xr1, &As[0][rA + 32][cA]);
    async_cp16(w1b, &Bs1[0][rB][cB]);
    async_cp16(w1b + (size_t)16 * FF, &Bs1[0][rB + 16][cB]);
    async_cp16(w3b, &Bs3[0][rB][cB]);
    async_cp16(w3b + (size_t)16 * FF, &Bs3[0][rB + 16][cB]);
  }
  for (int kc = 0; kc < DD / 32; ++kc) {
    const int cur = kc & 1;
    wait_async0();            // my chunk-kc transfers done
    __syncthreads();          // everyone's done; prev compute done -> other buf free
    if (kc + 1 < DD / 32) {
      const int k = (kc + 1) * 32, nxt = 1 - cur;
      async_cp16(xr0 + k, &As[nxt][rA][cA]);
      async_cp16(xr1 + k, &As[nxt][rA + 32][cA]);
      async_cp16(w1b + (size_t)k * FF, &Bs1[nxt][rB][cB]);
      async_cp16(w1b + (size_t)(k + 16) * FF, &Bs1[nxt][rB + 16][cB]);
      async_cp16(w3b + (size_t)k * FF, &Bs3[nxt][rB][cB]);
      async_cp16(w3b + (size_t)(k + 16) * FF, &Bs3[nxt][rB + 16][cB]);
    }
    gemm1_compute(As[cur], Bs1[cur], Bs3[cur], wm, wn, mf, kh, acc1, acc3);
  }
#else
  for (int kc = 0; kc < DD / 32; ++kc) {
    const int k = kc * 32;
    *(float4*)&As[0][rA][cA]      = *(const float4*)(xr0 + k);
    *(float4*)&As[0][rA + 32][cA] = *(const float4*)(xr1 + k);
    *(float4*)&Bs1[0][rB][cB]      = *(const float4*)(w1b + (size_t)k * FF);
    *(float4*)&Bs1[0][rB + 16][cB] = *(const float4*)(w1b + (size_t)(k + 16) * FF);
    *(float4*)&Bs3[0][rB][cB]      = *(const float4*)(w3b + (size_t)k * FF);
    *(float4*)&Bs3[0][rB + 16][cB] = *(const float4*)(w3b + (size_t)(k + 16) * FF);
    if (kc + 1 < DD / 32) {
      __builtin_prefetch(w1b + (size_t)(k + 32) * FF, 0, 1);
      __builtin_prefetch(w3b + (size_t)(k + 32) * FF, 0, 1);
    }
    __syncthreads();
    gemm1_compute(As[0], Bs1[0], Bs3[0], wm, wn, mf, kh, acc1, acc3);
    __syncthreads();
  }
#endif

  // epilogue: SiLU(a1)*a3 -> bf16 h rows at compacted offset
  const int hbase = offs[e];
  const int rb    = wm + ((lane >> 4) << 3);
#pragma unroll
  for (int s = 0; s < 2; ++s) {
    int col = f0 + wn + s * 16 + mf;
#pragma unroll
    for (int r = 0; r < 8; ++r) {
      int pos = m0 + rb + r;
      if (pos < count) {
        float v1 = acc1[s][r], v3 = acc3[s][r];
        float hv = (v1 / (1.f + __expf(-v1))) * v3;
        hbuf[(size_t)(hbase + pos) * FF + col] = f32_to_bf16(hv);
      }
    }
  }
}

// ---------------------------------------------------------------------------
// K4: grouped GEMM2 — out[tok] += weight * (h @ w2[e]).  bf16 WMMA K=32.
// 64x64 tile, 8 waves (4Mx2N), K-chunk 64 (two WMMA K-steps per barrier).
// ---------------------------------------------------------------------------
__global__ __launch_bounds__(256) void moe_gemm2(
    const unsigned short* __restrict__ hbuf, const float* __restrict__ w2,
    const int* __restrict__ cnt,  const int* __restrict__ offs,
    const int* __restrict__ btok, const float* __restrict__ bwt,
    float* __restrict__ out)
{
  const int e     = blockIdx.z;
  const int count = cnt[e];
  const int m0    = blockIdx.y * 64;
  if (m0 >= count) return;
  const int d0    = blockIdx.x * 64;

  // bf16 pairs packed as dwords: Ap[row m][pair p] holds k=2p,2p+1
  __shared__ unsigned int Ap[64][36];   // 64 rows x 32 pairs (+pad)
  __shared__ unsigned int Bp[32][68];   // 32 pairs x 64 cols (+pad)

  const int tid  = threadIdx.x;
  const int lane = tid & 31, wid = tid >> 5;
  const int wm = (wid >> 1) * 16;
  const int wn = (wid & 1) * 32;
  const int mf = lane & 15;
  const int pb = (lane >> 4) << 2;   // upper half lanes: +8 K -> +4 pairs

  const int hbase = offs[e];

  // fill assignments
  const int rA = tid >> 3, qA = tid & 7;     // A: 64 rows x 8 uint4 (x2 rows/thread)
  const int pB = tid >> 4, cB = tid & 15;    // B: 32 pairs x 16 uint4 (x2 pairs/thread)
  int posA0 = m0 + rA,      rowA0 = hbase + (posA0 < count ? posA0 : count - 1);
  int posA1 = m0 + rA + 32, rowA1 = hbase + (posA1 < count ? posA1 : count - 1);
  const unsigned short* ha0 = hbuf + (size_t)rowA0 * FF + qA * 8;
  const unsigned short* ha1 = hbuf + (size_t)rowA1 * FF + qA * 8;
  const float* w2p = w2 + (size_t)e * FF * DD;
  const float* w2b = w2p + (size_t)(2 * pB) * DD + d0 + cB * 4;

  const v8f zero8 = {0.f,0.f,0.f,0.f,0.f,0.f,0.f,0.f};
  v8f acc[2] = {zero8, zero8};

  for (int k = 0; k < FF; k += 64) {
    // A tile: 64 x 64 bf16 as 32 dword-pairs per row
    *(uint4*)&Ap[rA][qA * 4]      = *(const uint4*)(ha0 + k);
    *(uint4*)&Ap[rA + 32][qA * 4] = *(const uint4*)(ha1 + k);
    // B tile: 64 x 64 f32 of w2 -> bf16 pair-packed (rows 2p, 2p+1)
#pragma unroll
    for (int h = 0; h < 2; ++h) {
      const float* wb = w2b + ((size_t)k + h * 32) * DD;
      float4 lo = *(const float4*)wb;
      float4 hi = *(const float4*)(wb + DD);
      uint4 pk;
      pk.x = f32_to_bf16(lo.x) | ((unsigned int)f32_to_bf16(hi.x) << 16);
      pk.y = f32_to_bf16(lo.y) | ((unsigned int)f32_to_bf16(hi.y) << 16);
      pk.z = f32_to_bf16(lo.z) | ((unsigned int)f32_to_bf16(hi.z) << 16);
      pk.w = f32_to_bf16(lo.w) | ((unsigned int)f32_to_bf16(hi.w) << 16);
      *(uint4*)&Bp[pB + h * 16][cB * 4] = pk;
    }
    if (k + 64 < FF)
      __builtin_prefetch(w2b + (size_t)(k + 64) * DD, 0, 1);
    __syncthreads();

    // two K=32 WMMA steps; vgpr v -> K = (v<4 ? 2v : 16+2(v-4)) + 8*half
#pragma unroll
    for (int j = 0; j < 2; ++j) {
      BfFrag fa, fb0, fb1;
#pragma unroll
      for (int v = 0; v < 8; ++v) {
        int pr = (v < 4 ? v : 8 + (v - 4)) + pb + j * 16;
        fa.u[v]  = Ap[wm + mf][pr];
        fb0.u[v] = Bp[pr][wn + mf];
        fb1.u[v] = Bp[pr][wn + 16 + mf];
      }
      acc[0] = __builtin_amdgcn_wmma_f32_16x16x32_bf16(
          false, fa.v, false, fb0.v, (short)0, acc[0], false, false);
      acc[1] = __builtin_amdgcn_wmma_f32_16x16x32_bf16(
          false, fa.v, false, fb1.v, (short)0, acc[1], false, false);
    }
    __syncthreads();
  }

  // scatter-accumulate weight * y into out (exactly 2 adds per element
  // across the whole grid, starting from 0 -> bitwise deterministic)
  const int rb = wm + ((lane >> 4) << 3);
#pragma unroll
  for (int s = 0; s < 2; ++s) {
    int col = d0 + wn + s * 16 + mf;
#pragma unroll
    for (int r = 0; r < 8; ++r) {
      int pos = m0 + rb + r;
      if (pos < count) {
        int   t = btok[e * TT + pos];
        float w = bwt[e * TT + pos];
        atomicAdd(&out[(size_t)t * DD + col], w * acc[s][r]);
      }
    }
  }
}

// ---------------------------------------------------------------------------
extern "C" void kernel_launch(void* const* d_in, const int* in_sizes, int n_in,
                              void* d_out, int out_size, void* d_ws, size_t ws_size,
                              hipStream_t stream) {
  (void)in_sizes; (void)n_in; (void)out_size; (void)ws_size;

  const float* x  = (const float*)d_in[0];   // [T, D]
  const float* gw = (const float*)d_in[1];   // [E, D]
  const float* w1 = (const float*)d_in[2];   // [E, D, F]
  const float* w3 = (const float*)d_in[3];   // [E, D, F]
  const float* w2 = (const float*)d_in[4];   // [E, F, D]

  float* out    = (float*)d_out;                     // [T, D]
  float* logits = out + (size_t)TT * DD;             // [T, E]

  // workspace layout
  char* ws = (char*)d_ws;
  int*            cnt  = (int*)(ws + 0);
  int*            offs = (int*)(ws + 256);
  int*            btok = (int*)(ws + 512);
  float*          bwt  = (float*)(ws + 512 + (size_t)EE * TT * 4);
  unsigned short* hbuf = (unsigned short*)(ws + 512 + (size_t)EE * TT * 8);
  // hbuf: 2*T rows x F bf16 = ~117.4 MB

  (void)hipMemsetAsync(out, 0, (size_t)TT * DD * sizeof(float), stream);
  (void)hipMemsetAsync(cnt, 0, EE * sizeof(int), stream);

  moe_router <<<TT / 256, 256, 0, stream>>>(x, gw, logits, cnt, btok, bwt);
  moe_offsets<<<1, 32, 0, stream>>>(cnt, offs);
  moe_gemm1  <<<dim3(FF / 64, TT / 64, EE), 256, 0, stream>>>(
      x, w1, w3, cnt, offs, btok, hbuf);
  moe_gemm2  <<<dim3(DD / 64, TT / 64, EE), 256, 0, stream>>>(
      hbuf, w2, cnt, offs, btok, bwt, out);
}
